// MultiHead_23338852286555
// MI455X (gfx1250) — matrix operand order repaired and checked by reading, not compiled, not run on hardware
//
#include <hip/hip_runtime.h>
#include <hip/hip_bf16.h>
#include <math.h>

// ---------------------------------------------------------------------------
// MI455X (gfx1250) fused transformer layer.
// f16 WMMA (v_wmma_f32_16x16x32_f16), f32 accumulation, software-pipelined
// B-fragment prefetch, TDM (tensor_load_to_lds) staging for FFN A-tiles.
// ---------------------------------------------------------------------------

typedef __attribute__((ext_vector_type(16))) _Float16 v16h;
typedef __attribute__((ext_vector_type(8)))  _Float16 v8h;
typedef __attribute__((ext_vector_type(8)))  float    v8f;
typedef __attribute__((ext_vector_type(4)))  unsigned int u32x4;
typedef __attribute__((ext_vector_type(8)))  int      i32x8;
typedef __attribute__((ext_vector_type(4)))  int      i32x4;

#define Bdim 4
#define Ldim 2048
#define Ddim 512
#define Edim 512
#define Hnum 8
#define DH   64
#define BL   (Bdim * Ldim)   // 8192 rows

__device__ __forceinline__ v8f wmma16(v16h a, v16h b, v8f c) {
  return __builtin_amdgcn_wmma_f32_16x16x32_f16(false, a, false, b,
                                                (short)0, c, false, false);
}

// A-fragment (16x32 f16) from row-major [.., ld]: lane m=lane&15 holds row m;
// lanes<16 K in {0..7,16..23}, lanes>=16 K in {8..15,24..31}.
__device__ __forceinline__ v16h frag_a(const _Float16* p, int ld, int lane) {
  int m  = lane & 15;
  int kb = (lane & 16) ? 8 : 0;
  v8h g0 = *(const v8h*)(p + m * ld + kb);
  v8h g1 = *(const v8h*)(p + m * ld + kb + 16);
  v16h a;
#pragma unroll
  for (int e = 0; e < 8; ++e) { a[e] = g0[e]; a[e + 8] = g1[e]; }
  return a;
}

// B-fragment (32x16 f16) from K-contiguous Bt[n][k], stride ld: lane holds
// column n=lane&15; lanes<16 K in {0..7,16..23}, lanes>=16 K in {8..15,24..31}.
__device__ __forceinline__ v16h frag_b(const _Float16* p, int ld, int lane) {
  int n  = lane & 15;
  int kb = (lane & 16) ? 8 : 0;
  v8h g0 = *(const v8h*)(p + n * ld + kb);
  v8h g1 = *(const v8h*)(p + n * ld + kb + 16);
  v16h b;
#pragma unroll
  for (int e = 0; e < 8; ++e) { b[e] = g0[e]; b[e + 8] = g1[e]; }
  return b;
}

__device__ __forceinline__ float wave_sum32(float v) {
#pragma unroll
  for (int off = 16; off > 0; off >>= 1) v += __shfl_xor(v, off, 32);
  return v;
}
__device__ __forceinline__ float wave_max32(float v) {
#pragma unroll
  for (int off = 16; off > 0; off >>= 1) v = fmaxf(v, __shfl_xor(v, off, 32));
  return v;
}
__device__ __forceinline__ v8f vzero8() {
  v8f z = {0.f, 0.f, 0.f, 0.f, 0.f, 0.f, 0.f, 0.f};
  return z;
}

// ---------------------------------------------------------------------------
// TDM: 1-D contiguous f16 tile (nelem 2-byte elements) global -> LDS.
// D# built per CDNA5 ISA ch.8: group0 {count=1, lds_addr, global_addr, type=2},
// group1 {data_size=2B, tensor_dim0/tile_dim0 = nelem, stride = nelem}.
// Issue from ONE wave; pair with s_wait_tensorcnt(0) then workgroup barrier.
// ---------------------------------------------------------------------------
__device__ __forceinline__ void tdm_load_1d(unsigned lds_off, const void* gptr,
                                            unsigned nelem) {
  unsigned long long ga = (unsigned long long)gptr;
  u32x4 g0;
  g0[0] = 1u;                                       // count=1 (valid), user mode
  g0[1] = lds_off;                                  // lds_addr (bytes)
  g0[2] = (unsigned)(ga & 0xffffffffu);             // global_addr[31:0]
  g0[3] = (unsigned)((ga >> 32) & 0x1ffffffu)       // global_addr[56:32]
        | (2u << 30);                               // type = 2 ("image")
  i32x8 g1;
  g1[0] = (int)(1u << 16);                          // wg_mask=0, data_size=1 (2B)
  g1[1] = (int)((nelem & 0xffffu) << 16);           // tensor_dim0[15:0]
  g1[2] = (int)(((nelem >> 16) & 0xffffu)           // tensor_dim0[31:16]
        | (1u << 16));                              // tensor_dim1 = 1
  g1[3] = (int)((nelem & 0xffffu) << 16);           // tile_dim0 = nelem
  g1[4] = 0;                                        // tile_dim1/2 = 0 (unused)
  g1[5] = (int)nelem;                               // tensor_dim0_stride[31:0]
  g1[6] = 0;
  g1[7] = 0;
  i32x4 z4 = {0, 0, 0, 0};
  i32x8 z8 = {0, 0, 0, 0, 0, 0, 0, 0};
  __builtin_amdgcn_tensor_load_to_lds(g0, g1, z4, z4, z8, 0);
}

// ---------------------------------------------------------------------------
// Pack fp32 weight W[K][N] into f16 transposed Wt[N][K] (B-operand layout).
// ---------------------------------------------------------------------------
__global__ void k_w_pack(const float* __restrict__ W, _Float16* __restrict__ Wt,
                         int K, int N) {
  int idx = blockIdx.x * blockDim.x + threadIdx.x;
  if (idx < K * N) {
    int k = idx / N, n = idx % N;
    Wt[(size_t)n * K + k] = (_Float16)W[idx];
  }
}

// ---------------------------------------------------------------------------
// Fused: Y = LayerNorm(X @ W + b); scatter into per-head f16 layout.
// Block: 16 rows x 512 cols, 8 waves x 4 N-tiles. B frags double-buffered.
// ---------------------------------------------------------------------------
__global__ void k_qkv(const float* __restrict__ X, const _Float16* __restrict__ Wt,
                      const float* __restrict__ bias, const float* __restrict__ g,
                      const float* __restrict__ bn, _Float16* __restrict__ dst,
                      int vtrans) {
  __shared__ __align__(16) _Float16 sA[16 * 512];
  __shared__ __align__(16) float    sC[16 * 512];
  int tid = threadIdx.x, w = tid >> 5, lane = tid & 31;
  int row0 = blockIdx.x * 16;

  for (int i = tid; i < 16 * 512; i += 256)
    sA[i] = (_Float16)X[(size_t)(row0 + (i >> 9)) * 512 + (i & 511)];
  __syncthreads();

  const _Float16* wp = Wt + (size_t)(w * 64) * 512;
  v8f acc[4];
  v16h bc[4];
#pragma unroll
  for (int j = 0; j < 4; ++j) {
    acc[j] = vzero8();
    bc[j]  = frag_b(wp + (size_t)j * 16 * 512, 512, lane);
  }
#pragma unroll
  for (int ks = 0; ks < 15; ++ks) {                 // pipelined steps
    v16h a = frag_a(sA + ks * 32, 512, lane);
    v16h bn_[4];
#pragma unroll
    for (int j = 0; j < 4; ++j)
      bn_[j] = frag_b(wp + (size_t)j * 16 * 512 + (ks + 1) * 32, 512, lane);
#pragma unroll
    for (int j = 0; j < 4; ++j) acc[j] = wmma16(a, bc[j], acc[j]);
#pragma unroll
    for (int j = 0; j < 4; ++j) bc[j] = bn_[j];
  }
  {                                                 // epilogue step ks=15
    v16h a = frag_a(sA + 15 * 32, 512, lane);
#pragma unroll
    for (int j = 0; j < 4; ++j) acc[j] = wmma16(a, bc[j], acc[j]);
  }

  int n0 = lane & 15, mo = (lane & 16) ? 8 : 0;
#pragma unroll
  for (int j = 0; j < 4; ++j) {
    int col = w * 64 + j * 16 + n0;
    float bv = bias[col];
#pragma unroll
    for (int r = 0; r < 8; ++r) sC[(mo + r) * 512 + col] = acc[j][r] + bv;
  }
  __syncthreads();

#pragma unroll
  for (int rr = 0; rr < 2; ++rr) {
    int row = w * 2 + rr;
    float s = 0.f, sq = 0.f;
#pragma unroll
    for (int i = 0; i < 16; ++i) {
      float x = sC[row * 512 + lane + i * 32];
      s += x; sq += x * x;
    }
    s = wave_sum32(s); sq = wave_sum32(sq);
    float mean = s * (1.f / 512.f);
    float var  = sq * (1.f / 512.f) - mean * mean;
    float rstd = rsqrtf(var + 1e-5f);
    int t = row0 + row, b = t >> 11, l = t & 2047;
#pragma unroll
    for (int i = 0; i < 16; ++i) {
      int col = lane + i * 32;
      float y = (sC[row * 512 + col] - mean) * rstd * g[col] + bn[col];
      int h = col >> 6, d = col & 63;
      size_t o = vtrans
        ? ((size_t)((b * Hnum + h) * DH + d) * Ldim + l)
        : ((size_t)((b * Hnum + h) * Ldim + l) * DH + d);
      dst[o] = (_Float16)y;
    }
  }
}

// ---------------------------------------------------------------------------
// Attention per (bh, 16 q-rows). grid = (128, 32), 256 threads.
// ---------------------------------------------------------------------------
__global__ void k_attn(const _Float16* __restrict__ q16,
                       const _Float16* __restrict__ k16,
                       const _Float16* __restrict__ vT,
                       float* __restrict__ cor, float* __restrict__ O) {
  __shared__ __align__(16) _Float16 sS[16 * Ldim];  // 64 KB; reused as f32 partials
  __shared__ float sMax[16];
  __shared__ float sSum[16];

  int tid = threadIdx.x, w = tid >> 5, lane = tid & 31;
  int bh = blockIdx.y;
  int qb = blockIdx.x * 16;
  const _Float16* qp = q16 + ((size_t)bh * Ldim + qb) * DH;
  const _Float16* kp = k16 + (size_t)bh * Ldim * DH;
  const _Float16* vp = vT + (size_t)bh * DH * Ldim;
  float* corp = cor + (size_t)bh * Ldim * Ldim + (size_t)qb * Ldim;
  int n0 = lane & 15, mo = (lane & 16) ? 8 : 0;

  v16h aq0 = frag_a(qp, DH, lane);
  v16h aq1 = frag_a(qp + 32, DH, lane);

  // ---- Phase 1: S = qk^T/8; CLS mask; -> cor (f32) + sS (f16). Pipelined. ----
  {
    int c0 = w * 16;
    v16h b0 = frag_b(kp + (size_t)c0 * DH, DH, lane);
    v16h b1 = frag_b(kp + (size_t)c0 * DH + 32, DH, lane);
    for (int it = 0; it < 16; ++it) {
      int c0n = c0 + 128;
      v16h p0 = b0, p1 = b1;
      if (it < 15) {
        b0 = frag_b(kp + (size_t)c0n * DH, DH, lane);
        b1 = frag_b(kp + (size_t)c0n * DH + 32, DH, lane);
      }
      v8f acc = vzero8();
      acc = wmma16(aq0, p0, acc);
      acc = wmma16(aq1, p1, acc);
#pragma unroll
      for (int r = 0; r < 8; ++r) {
        float sv = acc[r] * 0.125f;                 // 1/sqrt(dh)
        if ((c0 + n0) == 0) sv = -__builtin_inff(); // IS_CLS mask, column 0
        int m = mo + r;
        corp[(size_t)m * Ldim + c0 + n0] = sv;
        sS[m * Ldim + c0 + n0] = (_Float16)sv;
      }
      c0 = c0n;
    }
  }
  __syncthreads();

  // ---- Phase 2: row max + sum(exp); each wave owns 2 rows ----
#pragma unroll
  for (int rr = 0; rr < 2; ++rr) {
    int row = w * 2 + rr;
    float mx = -__builtin_inff();
    for (int i = 0; i < 64; ++i)
      mx = fmaxf(mx, (float)sS[row * Ldim + lane + i * 32]);
    mx = wave_max32(mx);
    float sum = 0.f;
    for (int i = 0; i < 64; ++i)
      sum += __expf((float)sS[row * Ldim + lane + i * 32] - mx);
    sum = wave_sum32(sum);
    if (lane == 0) { sMax[row] = mx; sSum[row] = sum; }
  }
  __syncthreads();

  // ---- Phase 3: O = P V, split-K across 8 waves, V frags double-buffered ----
  float rowMax = sMax[lane & 15];
  v8f o[4];
#pragma unroll
  for (int j = 0; j < 4; ++j) o[j] = vzero8();
  int kbA = (lane & 16) ? 8 : 0;
  int k0 = w * 8 * 32;
  v16h vb[4];
#pragma unroll
  for (int j = 0; j < 4; ++j)
    vb[j] = frag_b(vp + (size_t)j * 16 * Ldim + k0, Ldim, lane);
  for (int kk = 0; kk < 8; ++kk) {
    const _Float16* sp = &sS[(lane & 15) * Ldim + k0 + kbA];
    v16h pa;
#pragma unroll
    for (int e = 0; e < 8; ++e) {
      pa[e]     = (_Float16)__expf((float)sp[e] - rowMax);
      pa[e + 8] = (_Float16)__expf((float)sp[e + 16] - rowMax);
    }
    v16h cur[4];
#pragma unroll
    for (int j = 0; j < 4; ++j) cur[j] = vb[j];
    if (kk < 7) {
#pragma unroll
      for (int j = 0; j < 4; ++j)
        vb[j] = frag_b(vp + (size_t)j * 16 * Ldim + k0 + 32, Ldim, lane);
    }
#pragma unroll
    for (int j = 0; j < 4; ++j) o[j] = wmma16(pa, cur[j], o[j]);
    k0 += 32;
  }
  __syncthreads();      // all waves done with sS -> reuse as f32 partials

  float* sP = reinterpret_cast<float*>(sS);         // [8][16][64] = 32 KB
#pragma unroll
  for (int j = 0; j < 4; ++j)
#pragma unroll
    for (int r = 0; r < 8; ++r)
      sP[(w * 16 + mo + r) * 64 + j * 16 + n0] = o[j][r];
  __syncthreads();

  int b = bh >> 3, h = bh & 7;
  for (int i = tid; i < 16 * 64; i += 256) {
    int M = i >> 6, d = i & 63;
    float s = 0.f;
#pragma unroll
    for (int ww = 0; ww < 8; ++ww) s += sP[(ww * 16 + M) * 64 + d];
    s /= sSum[M];
    O[(size_t)(b * Ldim + qb + M) * Edim + h * DH + d] = s;
  }
}

// ---------------------------------------------------------------------------
// LayerNorm rows of O (f32 [8192][512]) -> f16 X1. One wave per row.
// ---------------------------------------------------------------------------
__global__ void k_ln1(const float* __restrict__ O, const float* __restrict__ g,
                      const float* __restrict__ bn, _Float16* __restrict__ X1) {
  int tid = threadIdx.x, w = tid >> 5, lane = tid & 31;
  int row = blockIdx.x * 8 + w;
  const float* p = O + (size_t)row * 512;
  float s = 0.f, sq = 0.f;
#pragma unroll
  for (int i = 0; i < 16; ++i) {
    float x = p[lane + i * 32];
    s += x; sq += x * x;
  }
  s = wave_sum32(s); sq = wave_sum32(sq);
  float mean = s * (1.f / 512.f);
  float var  = sq * (1.f / 512.f) - mean * mean;
  float rstd = rsqrtf(var + 1e-5f);
#pragma unroll
  for (int i = 0; i < 16; ++i) {
    int col = lane + i * 32;
    float y = (p[col] - mean) * rstd * g[col] + bn[col];
    X1[(size_t)row * 512 + col] = (_Float16)y;
  }
}

// ---------------------------------------------------------------------------
// FFN GEMM: Y = leaky(A @ W + b) [+ residual, f32 out on FINAL layer].
// A-tile (16 x KDIM f16, fully contiguous) staged via TDM tensor_load_to_lds.
// B frags double-buffered from global (weights stay L2-resident).
// ---------------------------------------------------------------------------
template <int KDIM, int NTW, bool FINAL>
__global__ void k_ffn(const _Float16* __restrict__ A, const _Float16* __restrict__ Bt,
                      const float* __restrict__ bias, const float* __restrict__ Res,
                      _Float16* __restrict__ Oh, float* __restrict__ Of) {
  constexpr int NOUT = NTW * 8 * 16;
  constexpr int KS   = KDIM / 32;
  __shared__ __align__(16) _Float16 sA[16 * KDIM];
  int tid = threadIdx.x, w = tid >> 5, lane = tid & 31;
  int row0 = blockIdx.x * 16;

  if (tid < 32) {       // one wave drives the Tensor Data Mover
    tdm_load_1d((unsigned)(size_t)sA, A + (size_t)row0 * KDIM, 16 * KDIM);
    __builtin_amdgcn_s_wait_tensorcnt(0);
  }
  __syncthreads();

  const _Float16* wp = Bt + (size_t)(w * NTW * 16) * KDIM;
  v8f acc[NTW];
  v16h bc[NTW];
#pragma unroll
  for (int j = 0; j < NTW; ++j) {
    acc[j] = vzero8();
    bc[j]  = frag_b(wp + (size_t)j * 16 * KDIM, KDIM, lane);
  }
#pragma unroll
  for (int ks = 0; ks < KS - 1; ++ks) {
    v16h a = frag_a(sA + ks * 32, KDIM, lane);
    v16h bn_[NTW];
#pragma unroll
    for (int j = 0; j < NTW; ++j)
      bn_[j] = frag_b(wp + (size_t)j * 16 * KDIM + (ks + 1) * 32, KDIM, lane);
#pragma unroll
    for (int j = 0; j < NTW; ++j) acc[j] = wmma16(a, bc[j], acc[j]);
#pragma unroll
    for (int j = 0; j < NTW; ++j) bc[j] = bn_[j];
  }
  {
    v16h a = frag_a(sA + (KS - 1) * 32, KDIM, lane);
#pragma unroll
    for (int j = 0; j < NTW; ++j) acc[j] = wmma16(a, bc[j], acc[j]);
  }

  int n0 = lane & 15, mo = (lane & 16) ? 8 : 0;
#pragma unroll
  for (int j = 0; j < NTW; ++j) {
    int col = w * NTW * 16 + j * 16 + n0;
    float bv = bias[col];
#pragma unroll
    for (int r = 0; r < 8; ++r) {
      int m = mo + r;
      float v = acc[j][r] + bv;
      v = (v > 0.f) ? v : 0.2f * v;                 // LeakyReLU(0.2)
      if constexpr (FINAL) {
        size_t o = (size_t)(row0 + m) * NOUT + col;
        Of[o] = v + Res[o];                         // + clone_Q residual
      } else {
        Oh[(size_t)(row0 + m) * NOUT + col] = (_Float16)v;
      }
    }
  }
}

// ---------------------------------------------------------------------------
extern "C" void kernel_launch(void* const* d_in, const int* in_sizes, int n_in,
                              void* d_out, int out_size, void* d_ws, size_t ws_size,
                              hipStream_t stream) {
  const float* Q   = (const float*)d_in[0];
  const float* K   = (const float*)d_in[1];
  const float* V   = (const float*)d_in[2];
  const float* Wq  = (const float*)d_in[3];
  const float* bq  = (const float*)d_in[4];
  const float* Wk  = (const float*)d_in[5];
  const float* bk  = (const float*)d_in[6];
  const float* Wv  = (const float*)d_in[7];
  const float* bv  = (const float*)d_in[8];
  const float* ng  = (const float*)d_in[9];
  const float* nb  = (const float*)d_in[10];
  const float* n1g = (const float*)d_in[11];
  const float* n1b = (const float*)d_in[12];
  const float* W1  = (const float*)d_in[13];
  const float* bf1 = (const float*)d_in[14];
  const float* W2  = (const float*)d_in[15];
  const float* bf2 = (const float*)d_in[16];

  char* ws = (char*)d_ws;
  size_t off = 0;
  _Float16* wq_t = (_Float16*)(ws + off); off += (size_t)512 * 512 * 2;
  _Float16* wk_t = (_Float16*)(ws + off); off += (size_t)512 * 512 * 2;
  _Float16* wv_t = (_Float16*)(ws + off); off += (size_t)512 * 512 * 2;
  _Float16* w1_t = (_Float16*)(ws + off); off += (size_t)1024 * 512 * 2;
  _Float16* w2_t = (_Float16*)(ws + off); off += (size_t)512 * 1024 * 2;
  _Float16* q16  = (_Float16*)(ws + off); off += (size_t)BL * 512 * 2;   // 8 MB
  _Float16* k16  = (_Float16*)(ws + off); off += (size_t)BL * 512 * 2;   // 8 MB
  _Float16* v16T = (_Float16*)(ws + off); off += (size_t)BL * 512 * 2;   // 8 MB
  float*    Obuf = (float*)(ws + off);    off += (size_t)BL * 512 * 4;   // 16 MB
  _Float16* X1 = q16;          // reuse: q16 dead after attention (8 MB)
  _Float16* X2 = k16;          // reuse: k16+v16T contiguous (16 MB)

  float* out = (float*)d_out;                     // [B,L,D] final
  float* cor = out + (size_t)Bdim * Ldim * Ddim;  // [B,H,L,L] scores

  k_w_pack<<<(512 * 512 + 255) / 256, 256, 0, stream>>>(Wq, wq_t, 512, 512);
  k_w_pack<<<(512 * 512 + 255) / 256, 256, 0, stream>>>(Wk, wk_t, 512, 512);
  k_w_pack<<<(512 * 512 + 255) / 256, 256, 0, stream>>>(Wv, wv_t, 512, 512);
  k_w_pack<<<(512 * 1024 + 255) / 256, 256, 0, stream>>>(W1, w1_t, 512, 1024);
  k_w_pack<<<(1024 * 512 + 255) / 256, 256, 0, stream>>>(W2, w2_t, 1024, 512);

  k_qkv<<<BL / 16, 256, 0, stream>>>(Q, wq_t, bq, ng, nb, q16, 0);
  k_qkv<<<BL / 16, 256, 0, stream>>>(K, wk_t, bk, ng, nb, k16, 0);
  k_qkv<<<BL / 16, 256, 0, stream>>>(V, wv_t, bv, ng, nb, v16T, 1);

  k_attn<<<dim3(Ldim / 16, Bdim * Hnum), 256, 0, stream>>>(q16, k16, v16T, cor, Obuf);

  k_ln1<<<BL / 8, 256, 0, stream>>>(Obuf, n1g, n1b, X1);

  k_ffn<512, 8, false><<<BL / 16, 256, 0, stream>>>(X1, w1_t, bf1, nullptr, X2, nullptr);
  k_ffn<1024, 4, true><<<BL / 16, 256, 0, stream>>>(X2, w2_t, bf2, Q, nullptr, out);
}
